// AutoEncoderTucker_67740224193011
// MI455X (gfx1250) — matrix operand order repaired and checked by reading, not compiled, not run on hardware
//
#include <hip/hip_runtime.h>

// ---------------------------------------------------------------------------
// AutoEncoderTucker on MI455X (gfx1250): fully factored Kronecker/Tucker
// pipeline using v_wmma_f32_16x16x32_bf16 for every contraction, with
// GLOBAL_LOAD_ASYNC_TO_LDS_B128 (ASYNCcnt) for all bf16 LDS staging.
// N=2048, J=K=192, R1=256, R2=R3=48.
// ---------------------------------------------------------------------------

typedef unsigned short u16;
typedef __bf16 bf16;
typedef bf16  bf16x16 __attribute__((ext_vector_type(16)));
typedef float f32x8   __attribute__((ext_vector_type(8)));
typedef unsigned int u32x4 __attribute__((ext_vector_type(4)));

#define NJK 36864   // J*K
#define JJ  192
#define KK  192
#define R1  256
#define R2  48
#define R3  48
#define RR  2304    // R2*R3

union FragU { u32x4 q[2]; bf16x16 v; };

__device__ __forceinline__ u16 f2bf(float f) {          // f32 -> bf16 (RNE)
  unsigned u = __float_as_uint(f);
  u += 0x7FFFu + ((u >> 16) & 1u);
  return (u16)(u >> 16);
}

__device__ __forceinline__ f32x8 zero8() {
  f32x8 z = {0.f, 0.f, 0.f, 0.f, 0.f, 0.f, 0.f, 0.f};
  return z;
}

// --- CDNA5 async global->LDS copy (ASYNCcnt path) ---------------------------
// LDS aperture: flat addr[31:0] == LDS byte offset, so low 32 bits of a
// __shared__ pointer give the dsaddr VGPR value.
__device__ __forceinline__ unsigned lds_off_of(const void* p) {
  return (unsigned)(unsigned long long)p;
}
__device__ __forceinline__ void async_b128(unsigned lds_off, const void* gaddr) {
  asm volatile("global_load_async_to_lds_b128 %0, %1, off"
               :: "v"(lds_off), "v"(gaddr) : "memory");
}
__device__ __forceinline__ void wait_async0() {
  asm volatile("s_wait_asynccnt 0x0" ::: "memory");
}

// A-operand fragment: row-major A[M x K] tile at base; lane m=lane&15 holds
// K = kh*8+0..7 (elems 0..7) and K = 16+kh*8+0..7 (elems 8..15), kh=lane>>4.
__device__ __forceinline__ bf16x16 load_a(const u16* base, int stride, int lane) {
  const int m = lane & 15, kh = lane >> 4;
  const u16* p = base + m * stride + kh * 8;
  FragU f;
  f.q[0] = *(const u32x4*)(p);
  f.q[1] = *(const u32x4*)(p + 16);
  return f.v;
}

// B-operand fragment from row-major B^T [N x K]: lane n=lane&15 holds
// K = kh*8+0..7 (elems 0..7) and K = 16+kh*8+0..7 (elems 8..15), kh=lane>>4.
__device__ __forceinline__ bf16x16 load_b(const u16* base, int stride, int lane) {
  const int n = lane & 15, kh = lane >> 4;
  const u16* p = base + n * stride + kh * 8;
  FragU f;
  f.q[0] = *(const u32x4*)(p);
  f.q[1] = *(const u32x4*)(p + 16);
  return f.v;
}

__device__ __forceinline__ f32x8 wmma_bf16(bf16x16 a, bf16x16 b, f32x8 c) {
  return __builtin_amdgcn_wmma_f32_16x16x32_bf16(false, a, false, b,
                                                 (short)0, c, false, false);
}

// D tile (16x16 f32): lane holds col n=lane&15, rows m = v + 8*(lane>>4).
__device__ __forceinline__ void store_d_f32(float* base, int stride, int lane, f32x8 d) {
  const int n = lane & 15, hl = lane >> 4;
#pragma unroll
  for (int v = 0; v < 8; ++v) base[(v + 8 * hl) * stride + n] = d[v];
}
__device__ __forceinline__ void store_d_u16(u16* base, int stride, int lane, f32x8 d) {
  const int n = lane & 15, hl = lane >> 4;
#pragma unroll
  for (int v = 0; v < 8; ++v) base[(v + 8 * hl) * stride + n] = f2bf(d[v]);
}
__device__ __forceinline__ void store_d_u16_T(u16* base, int stride, int lane, f32x8 d) {
  const int n = lane & 15, hl = lane >> 4;
#pragma unroll
  for (int v = 0; v < 8; ++v) base[n * stride + (v + 8 * hl)] = f2bf(d[v]);
}

// ---------------------------------------------------------------------------
// Kernel 0: prep small operands (softmax B/C, relu(G)^T, Ginv^T, bf16 copies)
// ---------------------------------------------------------------------------
__global__ void __launch_bounds__(256)
prep_kernel(const float* B, const float* C, const float* G,
            const float* Binv, const float* Cinv, const float* Ginv,
            u16* smB, u16* smC, u16* bBinv, u16* bCinv,
            u16* GinvT, u16* reluGT) {
  const int tid = threadIdx.x;
  const int g = blockIdx.x * blockDim.x + tid;
  const int gs = gridDim.x * blockDim.x;

  for (int idx = g; idx < R1 * RR; idx += gs) {       // Ginv^T [r1][c]
    int r1 = idx / RR, c = idx - r1 * RR;
    GinvT[idx] = f2bf(Ginv[c * R1 + r1]);
  }
  for (int idx = g; idx < RR * R1; idx += gs) {       // relu(G)^T [c][r1]
    int c = idx / R1, r1 = idx - c * R1;
    float v = G[r1 * RR + c];
    reluGT[idx] = f2bf(v > 0.f ? v : 0.f);
  }
  for (int idx = g; idx < R2 * JJ; idx += gs) bBinv[idx] = f2bf(Binv[idx]);
  for (int idx = g; idx < R3 * KK; idx += gs) bCinv[idx] = f2bf(Cinv[idx]);

  if (blockIdx.x == 0 && tid < JJ) {                  // softmax rows of B
    const float* row = B + tid * R2;
    float m = row[0];
    for (int i = 1; i < R2; ++i) m = fmaxf(m, row[i]);
    float s = 0.f;
    for (int i = 0; i < R2; ++i) s += __expf(row[i] - m);
    float inv = 1.f / s;
    for (int i = 0; i < R2; ++i) smB[tid * R2 + i] = f2bf(__expf(row[i] - m) * inv);
  }
  if (blockIdx.x == 1 && tid < KK) {                  // softmax rows of C
    const float* row = C + tid * R3;
    float m = row[0];
    for (int i = 1; i < R3; ++i) m = fmaxf(m, row[i]);
    float s = 0.f;
    for (int i = 0; i < R3; ++i) s += __expf(row[i] - m);
    float inv = 1.f / s;
    for (int i = 0; i < R3; ++i) smC[tid * R3 + i] = f2bf(__expf(row[i] - m) * inv);
  }
}

// ---------------------------------------------------------------------------
// Kernel 1: per-sample encode.  Z1 = Xn @ Binv^T (192x48, stored transposed),
// Z2 = Cinv @ Z1 (48x48) -> ws as (n, r3*48+r2) bf16.
// LDS: sX 73728 + sB 18432 + sC 18432 + sZ1T 19200 = 129792 B
// ---------------------------------------------------------------------------
__global__ void __launch_bounds__(256)
enc12_kernel(const float* X, const u16* bBinv, const u16* bCinv, u16* Z2) {
  extern __shared__ char smem[];
  u16* sX   = (u16*)(smem);             // 192 x 192
  u16* sB   = (u16*)(smem + 73728);     // 48 x 192  (Binv row-major = Bt)
  u16* sC   = (u16*)(smem + 92160);     // 48 x 192  (Cinv row-major = A)
  u16* sZ1T = (u16*)(smem + 110592);    // 48 x 200  (Z1^T, padded stride)
  const int tid = threadIdx.x, lane = tid & 31, wave = tid >> 5;
  const int n = blockIdx.x;

  // Async-stage Binv / Cinv (bf16, contiguous) while we convert X below.
  {
    const unsigned ob = lds_off_of(sB), oc = lds_off_of(sC);
    for (int i = tid; i < (R2 * JJ * 2) / 16; i += 256) {   // 1152 x b128 each
      async_b128(ob + i * 16, (const char*)bBinv + i * 16);
      async_b128(oc + i * 16, (const char*)bCinv + i * 16);
    }
  }
  // Stage Xn (f32 -> bf16) into LDS (needs VGPR conversion; overlaps asyncs).
  const float4* X4 = (const float4*)(X + (size_t)n * NJK);
  for (int i = tid; i < NJK / 4; i += 256) {
    float4 f = X4[i];
    ushort4 h;
    h.x = f2bf(f.x); h.y = f2bf(f.y); h.z = f2bf(f.z); h.w = f2bf(f.w);
    *(ushort4*)(sX + 4 * i) = h;
  }
  wait_async0();
  __syncthreads();

  // Stage 1: Z1 = Xn(192x192) @ Binv^T -> sZ1T[r2][k]   (12x3 tiles, K=192)
  for (int t = wave; t < 36; t += 8) {
    int mt = t / 3, nt = t - mt * 3;
    f32x8 acc = zero8();
#pragma unroll
    for (int kk = 0; kk < 6; ++kk) {
      bf16x16 a = load_a(sX + mt * 16 * 192 + kk * 32, 192, lane);
      bf16x16 b = load_b(sB + nt * 16 * 192 + kk * 32, 192, lane);
      acc = wmma_bf16(a, b, acc);
    }
    store_d_u16_T(sZ1T + (nt * 16) * 200 + mt * 16, 200, lane, acc);
  }
  __syncthreads();

  // Stage 2: Z2 = Cinv(48x192) @ Z1 -> global (48x48 per sample, 3x3 tiles)
  u16* out = Z2 + (size_t)n * RR;
  for (int t = wave; t < 9; t += 8) {
    int mt = t / 3, nt = t - mt * 3;
    f32x8 acc = zero8();
#pragma unroll
    for (int kk = 0; kk < 6; ++kk) {
      bf16x16 a = load_a(sC + mt * 16 * 192 + kk * 32, 192, lane);
      bf16x16 b = load_b(sZ1T + nt * 16 * 200 + kk * 32, 200, lane);
      acc = wmma_bf16(a, b, acc);
    }
    store_d_u16(out + mt * 16 * 48 + nt * 16, 48, lane, acc);
  }
}

// ---------------------------------------------------------------------------
// Kernel 2: 16 samples/block.  A = Z2tile(16x2304) @ Ginv(2304x256),
// fused row-softmax, T = smA(16x256) @ relu(G)(256x2304), scattered to ws
// as (n, r2, r3) so decode needs no transpose.
// LDS: sA 73728 + sF 16384 + sSm 8192 + red 2048 = 100352 B
// ---------------------------------------------------------------------------
__global__ void __launch_bounds__(256)
enc34_kernel(const u16* Z2, const u16* GinvT, const u16* reluGT, u16* T) {
  extern __shared__ char smem[];
  u16*   sA  = (u16*)(smem);            // 16 x 2304 bf16
  float* sF  = (float*)(smem + 73728);  // 16 x 256 f32 scores
  u16*   sSm = (u16*)(smem + 90112);    // 16 x 256 bf16 softmax
  float* red = (float*)(smem + 98304);  // 2 x 16 x 16 reductions
  const int tid = threadIdx.x, lane = tid & 31, wave = tid >> 5;
  const int s0 = blockIdx.x * 16;

  // Async-stage the 16x2304 bf16 Z2 tile (73728 B contiguous).
  {
    const unsigned oa = lds_off_of(sA);
    const char* src = (const char*)(Z2 + (size_t)s0 * RR);
    for (int i = tid; i < (16 * RR * 2) / 16; i += 256)     // 4608 x b128
      async_b128(oa + i * 16, src + i * 16);
  }
  wait_async0();
  __syncthreads();

  // GEMM: A-scores = Z2tile @ Ginv  (M=16, N=256 -> 2 n-tiles/wave, K=2304)
  {
    const int nt0 = wave * 2, nt1 = wave * 2 + 1;
    f32x8 acc0 = zero8(), acc1 = zero8();
    for (int kk = 0; kk < 72; ++kk) {
      bf16x16 a  = load_a(sA + kk * 32, RR, lane);
      bf16x16 b0 = load_b(GinvT + (size_t)nt0 * 16 * RR + kk * 32, RR, lane);
      bf16x16 b1 = load_b(GinvT + (size_t)nt1 * 16 * RR + kk * 32, RR, lane);
      acc0 = wmma_bf16(a, b0, acc0);
      acc1 = wmma_bf16(a, b1, acc1);
    }
    store_d_f32(sF + nt0 * 16, 256, lane, acc0);
    store_d_f32(sF + nt1 * 16, 256, lane, acc1);
  }
  __syncthreads();

  // Row softmax over 256 cols: 16 threads per row, 16 cols each.
  {
    const int r = tid >> 4, c = tid & 15;
    float* row = sF + r * 256;
    float lm = -3.4e38f;
    for (int i = 0; i < 16; ++i) lm = fmaxf(lm, row[c * 16 + i]);
    red[r * 16 + c] = lm;
    __syncthreads();
    float rm = red[r * 16];
    for (int i = 1; i < 16; ++i) rm = fmaxf(rm, red[r * 16 + i]);
    float ls = 0.f;
    for (int i = 0; i < 16; ++i) {
      float e = __expf(row[c * 16 + i] - rm);
      row[c * 16 + i] = e;
      ls += e;
    }
    red[256 + r * 16 + c] = ls;
    __syncthreads();
    float rs = 0.f;
    for (int i = 0; i < 16; ++i) rs += red[256 + r * 16 + i];
    float inv = 1.f / rs;
    for (int i = 0; i < 16; ++i) sSm[r * 256 + c * 16 + i] = f2bf(row[c * 16 + i] * inv);
  }
  __syncthreads();

  // GEMM: T = smA(16x256) @ relu(G)(256x2304).  A-frags hoisted (K=256).
  {
    bf16x16 af[8];
#pragma unroll
    for (int kk = 0; kk < 8; ++kk) af[kk] = load_a(sSm + kk * 32, 256, lane);
    for (int nt = wave; nt < 144; nt += 8) {
      f32x8 acc = zero8();
#pragma unroll
      for (int kk = 0; kk < 8; ++kk) {
        bf16x16 b = load_b(reluGT + (size_t)nt * 16 * 256 + kk * 32, 256, lane);
        acc = wmma_bf16(af[kk], b, acc);
      }
      const int nn = lane & 15, hl = lane >> 4;
      const int c = nt * 16 + nn, r3 = c / 48, r2 = c - r3 * 48;
#pragma unroll
      for (int v = 0; v < 8; ++v)
        T[(size_t)(s0 + v + 8 * hl) * RR + r2 * 48 + r3] = f2bf(acc[v]);
    }
  }
}

// ---------------------------------------------------------------------------
// Kernel 3: per-sample decode.  U = smC(192x48) @ Tn(48x48),
// out_n = U(192x48) @ smB^T.  K=48 zero-padded to 64 (2 WMMA k-steps).
// LDS: sSmC 24576 + sSmB 24576 + sTn 6144 + sU 24576 = 79872 B
// ---------------------------------------------------------------------------
__global__ void __launch_bounds__(256)
dec56_kernel(const u16* T, const u16* smC, const u16* smB, float* out) {
  extern __shared__ char smem[];
  u16* sSmC = (u16*)(smem);             // 192 x 64 (A for stage 5)
  u16* sSmB = (u16*)(smem + 24576);     // 192 x 64 (Bt for stage 6)
  u16* sTn  = (u16*)(smem + 49152);     // 48 x 64  (Bt = Tn^T, already (r2,r3))
  u16* sU   = (u16*)(smem + 55296);     // 192 x 64 (stage-5 result / stage-6 A)
  const int tid = threadIdx.x, lane = tid & 31, wave = tid >> 5;
  const int n = blockIdx.x;

  // Async-stage the 48-column bodies (96 B per row = 6 x b128) into the
  // stride-64 (128 B) LDS rows.
  {
    const unsigned oc = lds_off_of(sSmC), ob = lds_off_of(sSmB);
    for (int i = tid; i < 192 * 6; i += 256) {
      int r = i / 6, c = i - r * 6;
      async_b128(oc + r * 128 + c * 16, (const char*)smC + r * 96 + c * 16);
      async_b128(ob + r * 128 + c * 16, (const char*)smB + r * 96 + c * 16);
    }
    const unsigned ot = lds_off_of(sTn);
    const char* gT = (const char*)(T + (size_t)n * RR);
    for (int i = tid; i < 48 * 6; i += 256) {
      int r = i / 6, c = i - r * 6;
      async_b128(ot + r * 128 + c * 16, gT + r * 96 + c * 16);
    }
  }
  // Zero-fill pad columns (bytes disjoint from async destinations) and sU.
  {
    u32x4 zq = {0u, 0u, 0u, 0u};
    for (int i = tid; i < 192 * 2; i += 256) {      // sSmC / sSmB pads
      int r = i >> 1, c2 = i & 1;
      *(u32x4*)((char*)sSmC + r * 128 + 96 + c2 * 16) = zq;
      *(u32x4*)((char*)sSmB + r * 128 + 96 + c2 * 16) = zq;
    }
    for (int i = tid; i < 48 * 2; i += 256) {       // sTn pads
      int r = i >> 1, c2 = i & 1;
      *(u32x4*)((char*)sTn + r * 128 + 96 + c2 * 16) = zq;
    }
    for (int i = tid; i < (192 * 64 * 2) / 16; i += 256)  // sU = 0
      *(u32x4*)((char*)sU + i * 16) = zq;
  }
  wait_async0();
  __syncthreads();

  // Stage 5: U = smC @ Tn  (12 x 3 tiles, K = 64 incl. pad)
  for (int t = wave; t < 36; t += 8) {
    int mt = t / 3, nt = t - mt * 3;
    f32x8 acc = zero8();
#pragma unroll
    for (int kk = 0; kk < 2; ++kk) {
      bf16x16 a = load_a(sSmC + mt * 16 * 64 + kk * 32, 64, lane);
      bf16x16 b = load_b(sTn + nt * 16 * 64 + kk * 32, 64, lane);
      acc = wmma_bf16(a, b, acc);
    }
    store_d_u16(sU + mt * 16 * 64 + nt * 16, 64, lane, acc);
  }
  __syncthreads();

  // Stage 6: out_n = U @ smB^T  (12 x 12 tiles, K = 64 incl. pad), f32 out.
  float* o = out + (size_t)n * NJK;
  for (int t = wave; t < 144; t += 8) {
    int mt = t / 12, nt = t - mt * 12;
    f32x8 acc = zero8();
#pragma unroll
    for (int kk = 0; kk < 2; ++kk) {
      bf16x16 a = load_a(sU + mt * 16 * 64 + kk * 32, 64, lane);
      bf16x16 b = load_b(sSmB + nt * 16 * 64 + kk * 32, 64, lane);
      acc = wmma_bf16(a, b, acc);
    }
    store_d_f32(o + mt * 16 * 192 + nt * 16, 192, lane, acc);
  }
}

// ---------------------------------------------------------------------------
// Host-side launch
// ---------------------------------------------------------------------------
extern "C" void kernel_launch(void* const* d_in, const int* in_sizes, int n_in,
                              void* d_out, int out_size, void* d_ws, size_t ws_size,
                              hipStream_t stream) {
  const float* X    = (const float*)d_in[0];
  const float* B    = (const float*)d_in[1];
  const float* C    = (const float*)d_in[2];
  const float* G    = (const float*)d_in[3];
  const float* Binv = (const float*)d_in[4];
  const float* Cinv = (const float*)d_in[5];
  const float* Ginv = (const float*)d_in[6];
  float* out = (float*)d_out;
  const int N = in_sizes[0] / NJK;   // 2048

  char* ws = (char*)d_ws;
  u16* smB    = (u16*)(ws + 0);                               // 192*48
  u16* smC    = (u16*)(ws + 18432);                           // 192*48
  u16* bBinv  = (u16*)(ws + 36864);                           // 48*192
  u16* bCinv  = (u16*)(ws + 55296);                           // 48*192
  u16* GinvT  = (u16*)(ws + 73728);                           // 256*2304
  u16* reluGT = (u16*)(ws + 1253376);                         // 2304*256
  u16* Z2     = (u16*)(ws + 2433024);                         // N*2304
  u16* Tm     = (u16*)(ws + 2433024 + (size_t)N * RR * 2);    // N*2304

  prep_kernel<<<64, 256, 0, stream>>>(B, C, G, Binv, Cinv, Ginv,
                                      smB, smC, bBinv, bCinv, GinvT, reluGT);
  enc12_kernel<<<N, 256, 129792, stream>>>(X, bBinv, bCinv, Z2);
  enc34_kernel<<<N / 16, 256, 100352, stream>>>(Z2, GinvT, reluGT, Tm);
  dec56_kernel<<<N, 256, 79872, stream>>>(Tm, smC, smB, out);
}
